// YemongTemporal_50714973831686
// MI455X (gfx1250) — matrix operand order repaired and checked by reading, not compiled, not run on hardware
//
#include <hip/hip_runtime.h>
#include <hip/hip_bf16.h>
#include <math.h>

#define USE_ASYNC_LDS 1   // 1: global_load_async_to_lds_b128 path (ASYNCcnt)
                          // 0: register staging path (LOADcnt + ds_store)

// ---------------------------------------------------------------------------
// Model dims
// ---------------------------------------------------------------------------
#define BSZ      8
#define TLEN     1024
#define NTOK     (BSZ * TLEN)        // 8192
#define DMODEL   512
#define DINNER   1024
#define DSTATE   16
#define DTRANK   32
#define DCONV    4
#define NLAYERS  4
#define INDIM    32
#define ACTDIM   4
#define TGTDIM   32
#define ACTEMB   128
#define DCAT     (DMODEL + ACTEMB)   // 640

typedef __bf16 v16bf __attribute__((ext_vector_type(16)));
typedef float  v8f   __attribute__((ext_vector_type(8)));

__device__ __forceinline__ unsigned short f2bf(float f) {
    unsigned u = __builtin_bit_cast(unsigned, f);
    unsigned r = u + 0x7FFFu + ((u >> 16) & 1u);   // round-to-nearest-even
    return (unsigned short)(r >> 16);
}
__device__ __forceinline__ float silu(float v) { return v / (1.f + __expf(-v)); }

// ---------------------------------------------------------------------------
// bf16 WMMA GEMM: C(MxN) = A(MxK) @ B(KxN) [+ bias] [+ C]
//   A: row-major bf16 (lda). Staged into triple-buffered LDS with
//      global_load_async_to_lds_b128 (ASYNCcnt path, decoupled from B loads).
//   B: pre-swizzled into WMMA fragment order (k_swz_b), streamed from global
//      as coalesced b128 loads (LOADcnt path; resident in 192MB L2).
// Block 256 thr = 8 waves. Block tile 128x64; wave tile 32x32 (2x2 of 16x16).
// K multiple of 32; M multiple of 128; N multiple of 16 (epilogue-masked).
// ---------------------------------------------------------------------------
#define BM 128
#define BN 64
#define KT 32
#define AST 40   // LDS row stride (bf16): 80B = 20 banks -> conflict-free, 16B aligned

__global__ __launch_bounds__(256) void gemm_bf16_wmma(
    const unsigned short* __restrict__ A, int lda,
    const uint4* __restrict__ Bsw,          // swizzled: [K/32][N/16][2][32][4] dwords
    float* __restrict__ C, int ldc,
    const float* __restrict__ bias,
    int M, int N, int K, int addC)
{
    __shared__ __align__(16) unsigned short As[3][BM * AST];

    const int tid  = threadIdx.x;
    const int lane = tid & 31;
    const int wave = tid >> 5;
    const int wm   = wave >> 1;          // 0..3
    const int wn   = wave & 1;           // 0..1
    const int m0   = blockIdx.y * BM;
    const int hl   = lane >> 4;          // half-wave select
    const int l16  = lane & 15;
    const int N16  = N >> 4;

    // A-staging decomposition: 512 x 16B chunks, 2 per thread
    const int srow0 = (tid + 0)   >> 2, sc0 = (tid & 3) << 3;         // elems
    const int srow1 = (tid + 256) >> 2, sc1 = sc0;

    v8f acc[2][2];
#pragma unroll
    for (int i = 0; i < 2; ++i)
#pragma unroll
        for (int j = 0; j < 2; ++j)
#pragma unroll
            for (int r = 0; r < 8; ++r) acc[i][j][r] = 0.f;

    const int nkt = K >> 5;

#if USE_ASYNC_LDS
    // async DMA of one 16B chunk per lane, LDS dest addressed per-lane
    auto issueA = [&](int kti) {
        int k0 = kti << 5;
        unsigned short* buf = As[kti % 3];
        unsigned l0 = (unsigned)(uintptr_t)(&buf[srow0 * AST + sc0]);
        unsigned l1 = (unsigned)(uintptr_t)(&buf[srow1 * AST + sc1]);
        unsigned long long g0 =
            (unsigned long long)(uintptr_t)(A + (size_t)(m0 + srow0) * lda + k0 + sc0);
        unsigned long long g1 =
            (unsigned long long)(uintptr_t)(A + (size_t)(m0 + srow1) * lda + k0 + sc1);
        asm volatile("global_load_async_to_lds_b128 %0, %1, off"
                     :: "v"(l0), "v"(g0) : "memory");
        asm volatile("global_load_async_to_lds_b128 %0, %1, off"
                     :: "v"(l1), "v"(g1) : "memory");
    };
    issueA(0);
#else
    uint4 aLd0 = *(const uint4*)(A + (size_t)(m0 + srow0) * lda + sc0);
    uint4 aLd1 = *(const uint4*)(A + (size_t)(m0 + srow1) * lda + sc1);
#endif

    for (int kt = 0; kt < nkt; ++kt) {
        // ---- B fragments first: LOADcnt path, independent of A staging ----
        union { v16bf v; uint4 q[2]; } bfrag[2];
#pragma unroll
        for (int j = 0; j < 2; ++j) {
            int n16g = blockIdx.x * 4 + wn * 2 + j;
            if (n16g >= N16) n16g = N16 - 1;            // branchless edge clamp
            const uint4* bp = Bsw + ((size_t)kt * N16 + n16g) * 64 + lane;
            bfrag[j].q[0] = bp[0];
            bfrag[j].q[1] = bp[32];
        }

#if USE_ASYNC_LDS
        if (kt + 1 < nkt) {
            issueA(kt + 1);                              // tile kt+1 in flight
            asm volatile("s_wait_asynccnt 2" ::: "memory");   // tile kt landed
        } else {
            asm volatile("s_wait_asynccnt 0" ::: "memory");
        }
        __syncthreads();
        const unsigned short* Ab = As[kt % 3];
#else
        unsigned short* Ab = As[kt % 3];
        *(uint4*)(&Ab[srow0 * AST + sc0]) = aLd0;
        *(uint4*)(&Ab[srow1 * AST + sc1]) = aLd1;
        if (kt + 1 < nkt) {
            int k0 = (kt + 1) << 5;
            aLd0 = *(const uint4*)(A + (size_t)(m0 + srow0) * lda + k0 + sc0);
            aLd1 = *(const uint4*)(A + (size_t)(m0 + srow1) * lda + k0 + sc1);
        }
        __syncthreads();
#endif

        union { v16bf v; uint4 q[2]; } afrag[2];
#pragma unroll
        for (int i = 0; i < 2; ++i) {
            const unsigned short* ap = &Ab[(wm * 32 + i * 16 + l16) * AST + (hl << 3)];
            afrag[i].q[0] = *(const uint4*)(ap);
            afrag[i].q[1] = *(const uint4*)(ap + 16);
        }
#pragma unroll
        for (int i = 0; i < 2; ++i)
#pragma unroll
            for (int j = 0; j < 2; ++j)
                acc[i][j] = __builtin_amdgcn_wmma_f32_16x16x32_bf16(
                    false, afrag[i].v, false, bfrag[j].v,
                    (short)0, acc[i][j], false, false);
#if !USE_ASYNC_LDS
        // double-buffer safety handled by single barrier (see round-1 notes)
#endif
    }

    // ---- epilogue: C rows = r + 8*(lane/16), cols = lane%16 ---------------
#pragma unroll
    for (int i = 0; i < 2; ++i) {
#pragma unroll
        for (int j = 0; j < 2; ++j) {
            int col = blockIdx.x * BN + wn * 32 + j * 16 + l16;
            if (col < N) {
                float bval = bias ? bias[col] : 0.f;
#pragma unroll
                for (int r = 0; r < 8; ++r) {
                    int rowg = m0 + wm * 32 + i * 16 + r + (hl << 3);
                    size_t o = (size_t)rowg * ldc + col;
                    float vv = acc[i][j][r] + bval;
                    if (addC) vv += C[o];
                    C[o] = vv;
                }
            }
        }
    }
}

// ---------------------------------------------------------------------------
// Weight swizzle: f32 row-major (KxN) -> bf16 WMMA-B fragment order.
// Output dword o: v4=o&3, lane=(o>>2)&31, c=(o>>7)&1, tile=o>>8;
//   n16=tile%N16, kt=tile/N16, v=c*4+v4, K0=kt*32+c*16+(lane>>4)*8+2*v4, Nc=n16*16+(lane&15)
//   dword = bf16(W[K0][Nc]) | bf16(W[K0+1][Nc]) << 16
// ---------------------------------------------------------------------------
__global__ void k_swz_b(const float* __restrict__ W, unsigned* __restrict__ out,
                        int K, int N) {
    int o = blockIdx.x * 256 + threadIdx.x;
    int total = (K >> 1) * N;
    if (o >= total) return;
    int N16  = N >> 4;
    int v4   = o & 3;
    int lane = (o >> 2) & 31;
    int c    = (o >> 7) & 1;
    int tile = o >> 8;
    int n16  = tile % N16;
    int kt   = tile / N16;
    int v    = c * 4 + v4;
    int K0   = kt * 32 + (c << 4) + ((lane >> 4) << 3) + (v4 << 1); (void)v;
    int Nc   = n16 * 16 + (lane & 15);
    unsigned lo = f2bf(W[(size_t)K0 * N + Nc]);
    unsigned hi = f2bf(W[(size_t)(K0 + 1) * N + Nc]);
    out[o] = lo | (hi << 16);
}

// ---------------------------------------------------------------------------
// Elementwise / fused kernels
// ---------------------------------------------------------------------------
__global__ void k_cvt_bf16(const float* __restrict__ src,
                           unsigned short* __restrict__ dst, int n) {
    int i = blockIdx.x * 256 + threadIdx.x;
    if (i < n) dst[i] = f2bf(src[i]);
}

// cat = [state@W_state+b, silu(pa@W_act+b)] -> bf16, one block per token
__global__ __launch_bounds__(256) void k_embed(
    const float* __restrict__ st, const float* __restrict__ pa,
    const float* __restrict__ Wst, const float* __restrict__ bst,
    const float* __restrict__ Wact, const float* __restrict__ bact,
    unsigned short* __restrict__ cat)
{
    int t = blockIdx.x;
    __shared__ float s[INDIM], a[ACTDIM];
    if (threadIdx.x < INDIM)  s[threadIdx.x] = st[(size_t)t * INDIM + threadIdx.x];
    if (threadIdx.x < ACTDIM) a[threadIdx.x] = pa[(size_t)t * ACTDIM + threadIdx.x];
    __syncthreads();
    for (int o = threadIdx.x; o < DCAT; o += 256) {
        float v;
        if (o < DMODEL) {
            v = bst[o];
#pragma unroll
            for (int k = 0; k < INDIM; ++k) v += s[k] * Wst[k * DMODEL + o];
        } else {
            int oa = o - DMODEL;
            v = bact[oa];
#pragma unroll
            for (int k = 0; k < ACTDIM; ++k) v += a[k] * Wact[k * ACTEMB + oa];
            v = silu(v);
        }
        cat[(size_t)t * DCAT + o] = f2bf(v);
    }
}

// x = silu(rmsnorm(pre, g)) ; one block (256 thr) per token
__global__ __launch_bounds__(256) void k_fuse_post(
    const float* __restrict__ pre, const float* __restrict__ g,
    float* __restrict__ x)
{
    int t = blockIdx.x;
    const float* p = pre + (size_t)t * DMODEL;
    float ss = 0.f;
    for (int c = threadIdx.x; c < DMODEL; c += 256) { float v = p[c]; ss += v * v; }
    __shared__ float red[8]; __shared__ float rinv;
    for (int off = 16; off; off >>= 1) ss += __shfl_down(ss, off, 32);
    if ((threadIdx.x & 31) == 0) red[threadIdx.x >> 5] = ss;
    __syncthreads();
    if (threadIdx.x == 0) {
        float tot = 0.f;
        for (int i = 0; i < 8; ++i) tot += red[i];
        rinv = rsqrtf(tot / (float)DMODEL + 1e-6f);
    }
    __syncthreads();
    for (int c = threadIdx.x; c < DMODEL; c += 256) {
        float v = p[c] * g[c] * rinv;
        x[(size_t)t * DMODEL + c] = silu(v);
    }
}

// xn(bf16) = rmsnorm(x, w) ; one block per token
__global__ __launch_bounds__(256) void k_rmsnorm(
    const float* __restrict__ x, const float* __restrict__ w,
    unsigned short* __restrict__ xn)
{
    int t = blockIdx.x;
    const float* p = x + (size_t)t * DMODEL;
    float ss = 0.f;
    for (int c = threadIdx.x; c < DMODEL; c += 256) { float v = p[c]; ss += v * v; }
    __shared__ float red[8]; __shared__ float rinv;
    for (int off = 16; off; off >>= 1) ss += __shfl_down(ss, off, 32);
    if ((threadIdx.x & 31) == 0) red[threadIdx.x >> 5] = ss;
    __syncthreads();
    if (threadIdx.x == 0) {
        float tot = 0.f;
        for (int i = 0; i < 8; ++i) tot += red[i];
        rinv = rsqrtf(tot / (float)DMODEL + 1e-6f);
    }
    __syncthreads();
    for (int c = threadIdx.x; c < DMODEL; c += 256)
        xn[(size_t)t * DMODEL + c] = f2bf(p[c] * w[c] * rinv);
}

// causal depthwise conv (k=4) + silu over xz[:, :, 0:1024]
__global__ void k_conv(const float* __restrict__ xz,
                       const float* __restrict__ cw, const float* __restrict__ cb,
                       float* __restrict__ xc, unsigned short* __restrict__ xcb)
{
    size_t i = (size_t)blockIdx.x * 256 + threadIdx.x;     // over NTOK*DINNER
    if (i >= (size_t)NTOK * DINNER) return;
    int d = (int)(i & (DINNER - 1));
    size_t bt = i >> 10;
    int t = (int)(bt & (TLEN - 1));
    size_t brow = bt - t;
    float acc = cb[d];
#pragma unroll
    for (int k = 0; k < DCONV; ++k) {
        int tt = t - (DCONV - 1) + k;
        if (tt >= 0) acc += cw[d * DCONV + k] * xz[(brow + tt) * (2 * DINNER) + d];
    }
    float v = silu(acc);
    xc[i] = v;
    xcb[i] = f2bf(v);
}

// dt = softplus(xdbl[:, :32] @ W_dt + b_dt)
__global__ void k_dt(const float* __restrict__ xdbl,
                     const float* __restrict__ Wdt, const float* __restrict__ bdt,
                     float* __restrict__ dt)
{
    size_t i = (size_t)blockIdx.x * 256 + threadIdx.x;     // over NTOK*DINNER
    if (i >= (size_t)NTOK * DINNER) return;
    int d = (int)(i & (DINNER - 1));
    size_t bt = i >> 10;
    const float* xr = xdbl + bt * (DTRANK + 2 * DSTATE);
    float acc = bdt[d];
#pragma unroll
    for (int r = 0; r < DTRANK; ++r) acc += xr[r] * Wdt[r * DINNER + d];
    dt[i] = (acc > 20.f) ? acc : log1pf(__expf(acc));
}

// selective scan: one thread per (b, d) channel, 16 states in registers
__global__ __launch_bounds__(256) void k_scan(
    const float* __restrict__ dtb, const float* __restrict__ xc,
    const float* __restrict__ xdbl, const float* __restrict__ Alog,
    const float* __restrict__ Dp, float* __restrict__ y)
{
    int gid = blockIdx.x * 256 + threadIdx.x;              // 0..B*DINNER-1
    int b = gid >> 10, d = gid & (DINNER - 1);
    float A[DSTATE], h[DSTATE];
#pragma unroll
    for (int s = 0; s < DSTATE; ++s) { A[s] = -expf(Alog[d * DSTATE + s]); h[s] = 0.f; }
    float Dd = Dp[d];
    size_t base = (size_t)b * TLEN;
    for (int t = 0; t < TLEN; ++t) {
        size_t bt = base + t;
        float dtv = dtb[bt * DINNER + d];
        float xv  = xc[bt * DINNER + d];
        const float* xr = xdbl + bt * (DTRANK + 2 * DSTATE);
        float dx = dtv * xv;
        float yv = 0.f;
#pragma unroll
        for (int s = 0; s < DSTATE; ++s) {
            h[s] = __expf(dtv * A[s]) * h[s] + dx * xr[DTRANK + s];
            yv  += h[s] * xr[DTRANK + DSTATE + s];
        }
        y[bt * DINNER + d] = yv + xv * Dd;
    }
}

// ymul(bf16) = y * silu(z), z = xz[:, 1024:2048]
__global__ void k_gate(const float* __restrict__ y, const float* __restrict__ xz,
                       unsigned short* __restrict__ ymul)
{
    size_t i = (size_t)blockIdx.x * 256 + threadIdx.x;
    if (i >= (size_t)NTOK * DINNER) return;
    int d = (int)(i & (DINNER - 1));
    size_t bt = i >> 10;
    float z = xz[bt * (2 * DINNER) + DINNER + d];
    ymul[i] = f2bf(y[i] * silu(z));
}

__global__ void k_copy(const float* __restrict__ src, float* __restrict__ dst, int n) {
    int i = blockIdx.x * 256 + threadIdx.x;
    if (i < n) dst[i] = src[i];
}

// ---------------------------------------------------------------------------
// Host launch
// ---------------------------------------------------------------------------
static inline size_t alignup(size_t x) { return (x + 255) & ~(size_t)255; }

static void launch_gemm(const unsigned short* A, int lda, const void* Bsw,
                        float* C, int ldc, const float* bias,
                        int M, int N, int K, int addC, hipStream_t s)
{
    dim3 g((N + BN - 1) / BN, M / BM);
    gemm_bf16_wmma<<<g, 256, 0, s>>>(A, lda, (const uint4*)Bsw, C, ldc, bias,
                                     M, N, K, addC);
}

static void launch_swz(const float* W, void* out, int K, int N, hipStream_t s)
{
    int total = (K >> 1) * N;
    k_swz_b<<<(total + 255) / 256, 256, 0, s>>>(W, (unsigned*)out, K, N);
}

extern "C" void kernel_launch(void* const* d_in, const int* in_sizes, int n_in,
                              void* d_out, int out_size, void* d_ws, size_t ws_size,
                              hipStream_t stream)
{
    const float* state   = (const float*)d_in[0];
    const float* pact    = (const float*)d_in[1];
    const float* W_state = (const float*)d_in[2];
    const float* b_state = (const float*)d_in[3];
    const float* W_act   = (const float*)d_in[4];
    const float* b_act   = (const float*)d_in[5];
    const float* W_fuse  = (const float*)d_in[6];
    const float* b_fuse  = (const float*)d_in[7];
    const float* g_fuse  = (const float*)d_in[8];
    const float* norm_w  = (const float*)d_in[9];
    const float* W_in    = (const float*)d_in[10];
    const float* conv_w  = (const float*)d_in[11];
    const float* conv_b  = (const float*)d_in[12];
    const float* W_x     = (const float*)d_in[13];
    const float* W_dt    = (const float*)d_in[14];
    const float* b_dt    = (const float*)d_in[15];
    const float* A_log   = (const float*)d_in[16];
    const float* Dp      = (const float*)d_in[17];
    const float* W_out   = (const float*)d_in[18];
    const float* W_head  = (const float*)d_in[19];
    const float* b_head  = (const float*)d_in[20];

    char* ws = (char*)d_ws;
    size_t off = 0;
    auto carve = [&](size_t bytes) { char* p = ws + off; off += alignup(bytes); return p; };

    float* X    = (float*)carve((size_t)NTOK * DMODEL * 4);
    float* XZ   = (float*)carve((size_t)NTOK * 2 * DINNER * 4);   // also holds "pre"
    float* XC   = (float*)carve((size_t)NTOK * DINNER * 4);
    float* DT   = (float*)carve((size_t)NTOK * DINNER * 4);
    float* Y    = (float*)carve((size_t)NTOK * DINNER * 4);
    float* XDBL = (float*)carve((size_t)NTOK * (DTRANK + 2 * DSTATE) * 4);
    unsigned short* CATB = (unsigned short*)carve((size_t)NTOK * DCAT * 2);
    unsigned short* XNB  = (unsigned short*)carve((size_t)NTOK * DMODEL * 2);
    unsigned short* XCB  = (unsigned short*)carve((size_t)NTOK * DINNER * 2);
    unsigned short* YMB  = (unsigned short*)carve((size_t)NTOK * DINNER * 2);
    void* WFUSEB = carve((size_t)DCAT * DMODEL * 2);
    void* WINB   = carve((size_t)NLAYERS * DMODEL * 2 * DINNER * 2);
    void* WXB    = carve((size_t)NLAYERS * DINNER * (DTRANK + 2 * DSTATE) * 2);
    void* WOUTB  = carve((size_t)NLAYERS * DINNER * DMODEL * 2);
    void* WHEADB = carve((size_t)DMODEL * TGTDIM * 2);
    (void)ws_size; (void)in_sizes; (void)n_in; (void)out_size;

    // --- weight swizzle into WMMA fragment order (bf16) --------------------
    const int DXB = DTRANK + 2 * DSTATE;      // 64
    launch_swz(W_fuse, WFUSEB, DCAT, DMODEL, stream);
    launch_swz(W_head, WHEADB, DMODEL, TGTDIM, stream);
    for (int L = 0; L < NLAYERS; ++L) {
        launch_swz(W_in  + (size_t)L * DMODEL * 2 * DINNER,
                   (char*)WINB  + (size_t)L * DMODEL * 2 * DINNER * 2,
                   DMODEL, 2 * DINNER, stream);
        launch_swz(W_x   + (size_t)L * DINNER * DXB,
                   (char*)WXB   + (size_t)L * DINNER * DXB * 2,
                   DINNER, DXB, stream);
        launch_swz(W_out + (size_t)L * DINNER * DMODEL,
                   (char*)WOUTB + (size_t)L * DINNER * DMODEL * 2,
                   DINNER, DMODEL, stream);
    }

    // --- embed + fuse ------------------------------------------------------
    k_embed<<<NTOK, 256, 0, stream>>>(state, pact, W_state, b_state, W_act, b_act, CATB);
    launch_gemm(CATB, DCAT, WFUSEB, XZ, DMODEL, b_fuse, NTOK, DMODEL, DCAT, 0, stream);
    k_fuse_post<<<NTOK, 256, 0, stream>>>(XZ, g_fuse, X);

    // --- mamba layers ------------------------------------------------------
    const int NE = NTOK * DINNER;
    for (int L = 0; L < NLAYERS; ++L) {
        k_rmsnorm<<<NTOK, 256, 0, stream>>>(X, norm_w + (size_t)L * DMODEL, XNB);
        launch_gemm(XNB, DMODEL, (char*)WINB + (size_t)L * DMODEL * 2 * DINNER * 2,
                    XZ, 2 * DINNER, nullptr, NTOK, 2 * DINNER, DMODEL, 0, stream);
        k_conv<<<NE / 256, 256, 0, stream>>>(XZ, conv_w + (size_t)L * DINNER * DCONV,
                                             conv_b + (size_t)L * DINNER, XC, XCB);
        launch_gemm(XCB, DINNER, (char*)WXB + (size_t)L * DINNER * DXB * 2,
                    XDBL, DXB, nullptr, NTOK, DXB, DINNER, 0, stream);
        k_dt<<<NE / 256, 256, 0, stream>>>(XDBL, W_dt + (size_t)L * DTRANK * DINNER,
                                           b_dt + (size_t)L * DINNER, DT);
        k_scan<<<(BSZ * DINNER) / 256, 256, 0, stream>>>(
            DT, XC, XDBL, A_log + (size_t)L * DINNER * DSTATE, Dp + (size_t)L * DINNER, Y);
        k_gate<<<NE / 256, 256, 0, stream>>>(Y, XZ, YMB);
        launch_gemm(YMB, DINNER, (char*)WOUTB + (size_t)L * DINNER * DMODEL * 2,
                    X, DMODEL, nullptr, NTOK, DMODEL, DINNER, 1, stream);  // residual add
    }

    // --- head + output -----------------------------------------------------
    k_cvt_bf16<<<(NTOK * DMODEL + 255) / 256, 256, 0, stream>>>(X, XNB, NTOK * DMODEL);
    float* outp = (float*)d_out;
    launch_gemm(XNB, DMODEL, WHEADB, outp, TGTDIM, b_head, NTOK, TGTDIM, DMODEL, 0, stream);
    k_copy<<<(NTOK * DMODEL + 255) / 256, 256, 0, stream>>>(
        X, outp + (size_t)NTOK * TGTDIM, NTOK * DMODEL);
}
